// ContrastiveLoss_14001593385688
// MI455X (gfx1250) — matrix operand order repaired and checked by reading, not compiled, not run on hardware
//
#include <hip/hip_runtime.h>

// ---------------- problem constants (from reference) ----------------
#define INV_T     (1.0f / 0.07f)
#define EPS_REF   1e-8f

// D is fixed at 512 by the reference; LDS layout depends on it.
constexpr int KDIM   = 512;
constexpr int TILE_M = 128;
constexpr int TILE_N = 128;
constexpr int NSPLIT = 8;              // column splits across blocks
constexpr int LDA    = KDIM + 8;       // +16B pad -> lane stride 4 banks, conflict-free
constexpr size_t SMEM_BYTES =
    (size_t)(TILE_M + TILE_N) * LDA * sizeof(_Float16) + 2 * TILE_M * sizeof(float);

typedef _Float16 half8  __attribute__((ext_vector_type(8)));
typedef _Float16 v16h   __attribute__((ext_vector_type(16)));
typedef float    v8f    __attribute__((ext_vector_type(8)));
typedef float    f4     __attribute__((ext_vector_type(4)));

typedef unsigned int u32x4 __attribute__((ext_vector_type(4)));
typedef int          i32x4 __attribute__((ext_vector_type(4)));
typedef int          i32x8 __attribute__((ext_vector_type(8)));

#if __has_builtin(__builtin_amdgcn_tensor_load_to_lds) && \
    __has_builtin(__builtin_amdgcn_s_wait_tensorcnt)
#define HAVE_TDM 1
#else
#define HAVE_TDM 0
#endif

#if HAVE_TDM
// TDM load of one 128-row x 512-half tile (row-major, stride 512 halves in
// global) into LDS with a 16-byte pad after every 1024-byte row -> LDS row
// stride = 520 halves = LDA.  D# built per CDNA5 ISA #8.
//   group0: count=1 | lds_addr | global_addr | type=2
//   group1: data_size=2B, pad_enable, pad_interval=7(256 dw), pad_amount=3(4 dw),
//           tensor_dim0=512, tensor_dim1=128, tile_dim0=512, tile_dim1=128,
//           tensor_dim0_stride=512
__device__ __forceinline__ void tdm_load_tile(const _Float16* gsrc, void* ldst) {
    unsigned long long ga = (unsigned long long)(uintptr_t)gsrc;
    unsigned int       la = (unsigned int)(uintptr_t)ldst;  // low 32 bits = LDS offset
    u32x4 g0 = { 1u,                                   // count=1, no gather
                 la,                                   // lds_addr [63:32]
                 (unsigned int)(ga & 0xFFFFFFFFu),     // global_addr lo
                 ((unsigned int)((ga >> 32) & 0x01FFFFFFu)) | 0x80000000u }; // hi | type=2
    i32x8 g1 = { 0x07D10000,   // data_size=1(2B), pad_en, pad_interval=7, pad_amount=3
                 0x02000000,   // tensor_dim0 = 512 -> bits[63:48]
                 0x00800000,   // tensor_dim1 = 128 -> bits[95:80]
                 0x02000000,   // tile_dim0   = 512 -> bits[127:112]
                 0x00000080,   // tile_dim1   = 128 -> bits[143:128]
                 512,          // tensor_dim0_stride lo
                 0, 0 };
    i32x4 z4 = { 0, 0, 0, 0 };
#if defined(__clang_major__) && __clang_major__ >= 23
    i32x8 z8 = { 0, 0, 0, 0, 0, 0, 0, 0 };
    __builtin_amdgcn_tensor_load_to_lds(g0, g1, z4, z4, z8, 0);
#else
    __builtin_amdgcn_tensor_load_to_lds(g0, g1, z4, z4, 0);
#endif
}
#endif

// ---------------- kernel 0: zero the global accumulators ----------------
__global__ void scl_init_kernel(float* acc, int count) {
    int i = blockIdx.x * blockDim.x + threadIdx.x;
    if (i < count) acc[i] = 0.0f;
}

// ---------------- kernel 1: L2-normalize rows, f32 -> f16 ----------------
__global__ void scl_normalize_kernel(const float* __restrict__ E,
                                     _Float16* __restrict__ Eh, int N) {
    const int lane = threadIdx.x & 31;
    const int wid  = threadIdx.x >> 5;
    const int row  = blockIdx.x * 8 + wid;
    if (row >= N) return;

    const float* src = E + (size_t)row * KDIM + lane * 16;
    f4 a0 = *(const f4*)(src + 0);
    f4 a1 = *(const f4*)(src + 4);
    f4 a2 = *(const f4*)(src + 8);
    f4 a3 = *(const f4*)(src + 12);

    float ss = 0.0f;
#pragma unroll
    for (int j = 0; j < 4; ++j) ss += a0[j]*a0[j] + a1[j]*a1[j] + a2[j]*a2[j] + a3[j]*a3[j];
#pragma unroll
    for (int m = 1; m < 32; m <<= 1) ss += __shfl_xor(ss, m, 32);

    const float inv = rsqrtf(ss);
    half8 h0, h1;
#pragma unroll
    for (int j = 0; j < 4; ++j) {
        h0[j]     = (_Float16)(a0[j] * inv);
        h0[j + 4] = (_Float16)(a1[j] * inv);
        h1[j]     = (_Float16)(a2[j] * inv);
        h1[j + 4] = (_Float16)(a3[j] * inv);
    }
    _Float16* dst = Eh + (size_t)row * KDIM + lane * 16;
    *(half8*)(dst + 0) = h0;
    *(half8*)(dst + 8) = h1;
}

// ---------------- kernel 2: fused WMMA GEMM + masked exp row-reduction ----------------
// grid.x = N/TILE_M row tiles, grid.y = NSPLIT column splits; 512 threads = 16 waves.
// Wave (wm, wn): wm = wid>>2 in [0,4) -> 32 rows, wn = wid&3 in [0,4) -> 32 cols.
// Per wave: 2x2 WMMA tiles -> acc 32 VGPRs, frags 32 VGPRs (no spills).
__global__ void scl_gemm_kernel(const _Float16* __restrict__ Eh,
                                const long long* __restrict__ labels,
                                float* __restrict__ gNum, float* __restrict__ gDen,
                                int N) {
    extern __shared__ char smem[];
    _Float16* sA   = (_Float16*)smem;
    _Float16* sB   = sA + (size_t)TILE_M * LDA;
    float*    sNum = (float*)(sB + (size_t)TILE_N * LDA);
    float*    sDen = sNum + TILE_M;

    const int tid     = threadIdx.x;
    const int lane    = tid & 31;
    const int wid     = tid >> 5;
    const int wm      = wid >> 2;          // 0..3
    const int wn      = wid & 3;           // 0..3
    const int l16     = lane & 15;
    const int halfSel = lane >> 4;

    const int rowBase       = blockIdx.x * TILE_M;
    const int colSplit      = blockIdx.y * (N / NSPLIT);
    const int tilesPerSplit = (N / NSPLIT) / TILE_N;

    if (tid < TILE_M) { sNum[tid] = 0.0f; sDen[tid] = 0.0f; }

    // ---- Stage A rows once (TDM if available, else manual copy) ----
#if HAVE_TDM
    if (wid == 0) {
        tdm_load_tile(&Eh[(size_t)rowBase * KDIM], sA);
        __builtin_amdgcn_s_wait_tensorcnt(0);
    }
#else
    for (int idx = tid; idx < TILE_M * (KDIM / 8); idx += blockDim.x) {
        int r = idx >> 6, c = (idx & 63) * 8;
        *(half8*)&sA[r * LDA + c] = *(const half8*)&Eh[(size_t)(rowBase + r) * KDIM + c];
    }
#endif

    // Register-resident labels for this wave's 32 rows.
    int labM[2][8];
#pragma unroll
    for (int mi = 0; mi < 2; ++mi)
#pragma unroll
        for (int e = 0; e < 8; ++e)
            labM[mi][e] = (int)labels[rowBase + wm * 32 + mi * 16 + halfSel * 8 + e];

    __syncthreads();

    for (int ct = 0; ct < tilesPerSplit; ++ct) {
        const int colTileBase = colSplit + ct * TILE_N;

        // ---- Stage B tile ----
#if HAVE_TDM
        if (wid == 0) {
            tdm_load_tile(&Eh[(size_t)colTileBase * KDIM], sB);
            __builtin_amdgcn_s_wait_tensorcnt(0);
        }
#else
        for (int idx = tid; idx < TILE_N * (KDIM / 8); idx += blockDim.x) {
            int r = idx >> 6, c = (idx & 63) * 8;
            *(half8*)&sB[r * LDA + c] =
                *(const half8*)&Eh[(size_t)(colTileBase + r) * KDIM + c];
            if (ct + 1 < tilesPerSplit)
                __builtin_prefetch(&Eh[(size_t)(colTileBase + TILE_N + r) * KDIM + c], 0, 1);
        }
#endif
        __syncthreads();

        int labN[2];
#pragma unroll
        for (int ni = 0; ni < 2; ++ni)
            labN[ni] = (int)labels[colTileBase + wn * 32 + ni * 16 + l16];

        v8f acc[2][2];
#pragma unroll
        for (int mi = 0; mi < 2; ++mi)
#pragma unroll
            for (int ni = 0; ni < 2; ++ni) acc[mi][ni] = (v8f){};

        // K loop: 16 steps of K=32, 4 WMMAs per step per wave.
#pragma unroll 4
        for (int k0 = 0; k0 < KDIM; k0 += 32) {
            v16h aF[2], bF[2];
#pragma unroll
            for (int mi = 0; mi < 2; ++mi) {
                // A 16x32 f16 layout: lane row = l16; K strips [8h..8h+7], [16+8h..+7]
                const _Float16* pa =
                    &sA[(wm * 32 + mi * 16 + l16) * LDA + k0 + 8 * halfSel];
                half8 lo = *(const half8*)pa;
                half8 hi = *(const half8*)(pa + 16);
                aF[mi] = __builtin_shufflevector(lo, hi, 0,1,2,3,4,5,6,7,8,9,10,11,12,13,14,15);
            }
#pragma unroll
            for (int ni = 0; ni < 2; ++ni) {
                // B 32x16 f16 layout: lane col = l16; K strips [8h..8h+7], [16+8h..+7]
                const _Float16* pb =
                    &sB[(wn * 32 + ni * 16 + l16) * LDA + k0 + 8 * halfSel];
                half8 lo = *(const half8*)pb;
                half8 hi = *(const half8*)(pb + 16);
                bF[ni] = __builtin_shufflevector(lo, hi, 0,1,2,3,4,5,6,7,8,9,10,11,12,13,14,15);
            }
#pragma unroll
            for (int mi = 0; mi < 2; ++mi)
#pragma unroll
                for (int ni = 0; ni < 2; ++ni)
                    acc[mi][ni] = __builtin_amdgcn_wmma_f32_16x16x32_f16(
                        false, aF[mi], false, bF[ni], (short)0, acc[mi][ni], false, false);
        }

        // ---- Fused epilogue: exp + masks, reduce over this wave's 32 columns ----
#pragma unroll
        for (int mi = 0; mi < 2; ++mi) {
            float pn[8], pd[8];
#pragma unroll
            for (int e = 0; e < 8; ++e) { pn[e] = 0.0f; pd[e] = 0.0f; }
#pragma unroll
            for (int ni = 0; ni < 2; ++ni) {
                const int gn = colTileBase + wn * 32 + ni * 16 + l16;
#pragma unroll
                for (int e = 0; e < 8; ++e) {
                    const int gm = rowBase + wm * 32 + mi * 16 + halfSel * 8 + e;
                    const float s  = acc[mi][ni][e] * INV_T;
                    const float es = __expf(s);
                    const bool offDiag = (gm != gn);
                    if (offDiag) pd[e] += es;
                    if (offDiag && (labM[mi][e] == labN[ni]) && (s > 0.0f)) pn[e] += es;
                }
            }
            // Column reduction: xor 1..8 stays within each 16-lane half
            // (the two halves hold different row groups of the C layout).
#pragma unroll
            for (int e = 0; e < 8; ++e) {
#pragma unroll
                for (int m = 1; m < 16; m <<= 1) {
                    pn[e] += __shfl_xor(pn[e], m, 32);
                    pd[e] += __shfl_xor(pd[e], m, 32);
                }
            }
            if (l16 == 0) {
#pragma unroll
                for (int e = 0; e < 8; ++e) {
                    const int rl = wm * 32 + mi * 16 + halfSel * 8 + e;
                    atomicAdd(&sNum[rl], pn[e]);
                    atomicAdd(&sDen[rl], pd[e]);
                }
            }
        }
        __syncthreads();  // protect sB before restaging
    }

    // One global atomic per row per block (column splits combine here).
    if (tid < TILE_M) {
        atomicAdd(&gNum[rowBase + tid], sNum[tid]);
        atomicAdd(&gDen[rowBase + tid], sDen[tid]);
    }
}

// ---------------- kernel 3: per-row loss + global reduction ----------------
__global__ void scl_rowloss_kernel(const float* __restrict__ gNum,
                                   const float* __restrict__ gDen,
                                   float* gSum, float* gCnt, int N) {
    const int i = blockIdx.x * blockDim.x + threadIdx.x;
    float loss = 0.0f, cnt = 0.0f;
    if (i < N) {
        const float nu = gNum[i], de = gDen[i];
        if (nu > 0.0f && de > 0.0f) {
            loss = -logf(nu / (de + EPS_REF));
            cnt  = 1.0f;
        }
    }
#pragma unroll
    for (int m = 1; m < 32; m <<= 1) {
        loss += __shfl_xor(loss, m, 32);
        cnt  += __shfl_xor(cnt, m, 32);
    }
    if ((threadIdx.x & 31) == 0) {
        atomicAdd(gSum, loss);
        atomicAdd(gCnt, cnt);
    }
}

__global__ void scl_finalize_kernel(const float* gSum, const float* gCnt, float* out) {
    const float s = gSum[0], c = gCnt[0];
    const float mean = (c > 0.0f) ? (s / fmaxf(c, 1.0f)) : 0.0f;
    out[0] = fabsf(mean);
}

// ---------------- launcher ----------------
extern "C" void kernel_launch(void* const* d_in, const int* in_sizes, int n_in,
                              void* d_out, int out_size, void* d_ws, size_t ws_size,
                              hipStream_t stream) {
    const float*     emb    = (const float*)d_in[0];
    const long long* labels = (const long long*)d_in[1];
    float* out = (float*)d_out;

    const int N = in_sizes[1];            // 8192
    // workspace carve: Eh[N*KDIM] f16 | gNum[N] | gDen[N] | gSum | gCnt
    char* w = (char*)d_ws;
    _Float16* Eh  = (_Float16*)w;
    float* gNum = (float*)(w + (size_t)N * KDIM * sizeof(_Float16));
    float* gDen = gNum + N;
    float* gSum = gDen + N;
    float* gCnt = gSum + 1;

    const int accCount = 2 * N + 2;
    scl_init_kernel<<<(accCount + 255) / 256, 256, 0, stream>>>(gNum, accCount);

    scl_normalize_kernel<<<N / 8, 256, 0, stream>>>(emb, Eh, N);

    dim3 grid(N / TILE_M, NSPLIT);
    scl_gemm_kernel<<<grid, 512, SMEM_BYTES, stream>>>(Eh, labels, gNum, gDen, N);

    scl_rowloss_kernel<<<(N + 255) / 256, 256, 0, stream>>>(gNum, gDen, gSum, gCnt, N);
    scl_finalize_kernel<<<1, 1, 0, stream>>>(gSum, gCnt, out);
}
